// LSTMModel_3461743641123
// MI455X (gfx1250) — matrix operand order repaired and checked by reading, not compiled, not run on hardware
//
#include <hip/hip_runtime.h>
#include <hip/hip_bf16.h>

typedef __attribute__((ext_vector_type(16))) __bf16 v16bf;
typedef __attribute__((ext_vector_type(8)))  __bf16 bf16x8;
typedef __attribute__((ext_vector_type(8)))  float  v8f;

#define TLEN 200
#define ISZ  11
#define HSZ  100
#define GP   112     // padded per-gate width (7 * 16)
#define AROW 136     // padded A-tile row stride in bf16 elements (272B -> conflict-free)
#define BIASCOL 15   // A-tile column holding constant 1.0 (bias column)

__device__ __forceinline__ float fexp(float x)  { return __builtin_amdgcn_exp2f(x * 1.4426950408889634f); }
__device__ __forceinline__ float fsig(float x)  { return __builtin_amdgcn_rcpf(1.0f + fexp(-x)); }
__device__ __forceinline__ float ftanh(float x) { return 1.0f - 2.0f * __builtin_amdgcn_rcpf(fexp(2.0f * x) + 1.0f); }

// ---------------------------------------------------------------------------
// Prep: build pre-swizzled bf16 B-fragments of Wcat = [W_ih | pad,bias | W_hh | pad]
// (448 padded gate-outputs x 128 padded K). K column 15 carries (b_ih + b_hh),
// matched by a constant-1.0 column 15 in the A-tile -> bias folded into GEMM.
// Fragment f = ((g*7 + j)*4 + ks). Per fragment: 32 lanes x 16 bf16 (32B/lane,
// lane-contiguous). B k order chosen here (contiguous, same as A): lane = n (col),
// lanes 0-15 hold K = ks*32 + 0..15, lanes 16-31 hold K = ks*32 + 16..31.
// ---------------------------------------------------------------------------
__global__ void lstm_prep_kernel(const float* __restrict__ Wih,
                                 const float* __restrict__ Whh,
                                 const float* __restrict__ bih,
                                 const float* __restrict__ bhh,
                                 __bf16* __restrict__ wsW) {
    int tid = blockIdx.x * blockDim.x + threadIdx.x;
    if (tid < 112 * 512) {
        int f  = tid >> 9;            // fragment
        int l  = (tid >> 4) & 31;     // lane
        int e  = tid & 15;            // element within lane
        int ks = f & 3;
        int nt = f >> 2;
        int g  = nt / 7;
        int j  = nt % 7;
        int out = j * 16 + (l & 15);              // gate-local output index (0..111)
        int k   = ks * 32 + ((l >> 4) << 4) + e;  // K index (0..127)
        float v = 0.0f;
        if (out < HSZ) {
            int r = g * HSZ + out;                // row in original [400 x *] weights
            if (k < ISZ)                      v = Wih[r * ISZ + k];
            else if (k == BIASCOL)            v = bih[r] + bhh[r];
            else if (k >= 16 && k < 16 + HSZ) v = Whh[r * HSZ + (k - 16)];
        }
        wsW[tid] = (__bf16)v;
    }
}

// ---------------------------------------------------------------------------
// Main: one wave (32 threads) per block owning a 16-row batch stripe.
// c-state in registers; [16 x 128] bf16 A-tile ([x_t | 1 | h_{t-1}]) double-
// buffered in LDS; 7x4x4 = 112 v_wmma_f32_16x16x32_bf16 per timestep.
// First WMMA of each tile uses a shared zero C-octet (no accumulator-init
// copies). Weight fragments stream from global (hot in WGP$/L2); an asm
// memory barrier per timestep stops LICM from hoisting them into registers.
// ---------------------------------------------------------------------------
__global__ __launch_bounds__(32) void
lstm_main_kernel(const float* __restrict__ x,
                 const __bf16* __restrict__ wsW,
                 const float* __restrict__ W1, const float* __restrict__ b1,
                 const float* __restrict__ W2, const float* __restrict__ b2,
                 const float* __restrict__ W3, const float* __restrict__ b3,
                 float* __restrict__ out) {
    __shared__ __bf16 Ash[2][16 * AROW];

    const int lane = threadIdx.x & 31;
    const int rowBase = blockIdx.x * 16;          // global batch row of stripe
    const int half = lane >> 4;
    const int l15  = lane & 15;

    __bf16* Aw0 = &Ash[0][0];
    __bf16* Aw1 = &Ash[1][0];

    // Init both buffers: zeros, except constant-1.0 bias column.
    for (int e = lane; e < 16 * AROW; e += 32) {
        __bf16 v = ((e % AROW) == BIASCOL) ? (__bf16)1.0f : (__bf16)0.0f;
        Aw0[e] = v;
        Aw1[e] = v;
    }

    // Shared zero C-operand for the first WMMA of every tile (loop-invariant).
    v8f zf = {0.0f, 0.0f, 0.0f, 0.0f, 0.0f, 0.0f, 0.0f, 0.0f};

    // Cell state, C/D f32 tile layout: element v -> row m = v + 8*half, col n = lane&15.
    float cst[7][8];
#pragma unroll
    for (int j = 0; j < 7; ++j)
#pragma unroll
        for (int v = 0; v < 8; ++v) cst[j][v] = 0.0f;

    for (int t = 0; t < TLEN; ++t) {
        // Compiler-level memory barrier: keeps weight-fragment loads inside the
        // time loop (otherwise LICM hoists 112 fragments -> scratch spills).
        asm volatile("" ::: "memory");

        __bf16* Ac = (t & 1) ? Aw1 : Aw0;   // current [x_t | 1 | h_{t-1}]
        __bf16* An = (t & 1) ? Aw0 : Aw1;   // next (receives h_t)

        // Load x_t (16 rows x 11 f32) into current buffer cols 0..10.
        for (int e = lane; e < 16 * ISZ; e += 32) {
            int rl = e / ISZ, col = e % ISZ;
            float xv = x[((size_t)(rowBase + rl) * TLEN + (size_t)t) * ISZ + col];
            Ac[rl * AROW + col] = (__bf16)xv;
        }

        // Hoist the 4 distinct A-fragments for this timestep (shared by all j).
        // A k order used here (contiguous, same as B): lanes 0-15 K = ks*32 + 0..15, lanes 16-31 K = ks*32 + 16..31.
        v16bf af[4];
#pragma unroll
        for (int ks = 0; ks < 4; ++ks) {
            int ka = ks * 32 + half * 16;
            bf16x8 alo = *(const bf16x8*)(Ac + l15 * AROW + ka);
            bf16x8 ahi = *(const bf16x8*)(Ac + l15 * AROW + ka + 8);
            af[ks] = __builtin_shufflevector(alo, ahi,
                0,1,2,3,4,5,6,7,8,9,10,11,12,13,14,15);
        }

#pragma unroll
        for (int j = 0; j < 7; ++j) {
            v8f acc[4];
#pragma unroll
            for (int ks = 0; ks < 4; ++ks) {
                // Batch-load the 4 B-fragments of this k-step (8x b128, one clause).
                v16bf bm[4];
#pragma unroll
                for (int g = 0; g < 4; ++g) {
                    int frag = ((g * 7 + j) << 2) + ks;
                    const __bf16* bp = wsW + frag * 512 + lane * 16;  // 32B/lane
                    bf16x8 blo = *(const bf16x8*)(bp);
                    bf16x8 bhi = *(const bf16x8*)(bp + 8);
                    bm[g] = __builtin_shufflevector(blo, bhi,
                        0,1,2,3,4,5,6,7,8,9,10,11,12,13,14,15);
                }
#pragma unroll
                for (int g = 0; g < 4; ++g) {
                    acc[g] = __builtin_amdgcn_wmma_f32_16x16x32_bf16(
                        false, af[ks], false, bm[g], (short)0,
                        (ks == 0) ? zf : acc[g], false, false);
                }
            }
            // Elementwise LSTM cell update on the 16x16 tile.
#pragma unroll
            for (int v = 0; v < 8; ++v) {
                float ig = fsig(acc[0][v]);
                float fg = fsig(acc[1][v]);
                float gg = ftanh(acc[2][v]);
                float og = fsig(acc[3][v]);
                float cn = fg * cst[j][v] + ig * gg;
                cst[j][v] = cn;
                float hv = og * ftanh(cn);
                int m = v + half * 8;
                An[m * AROW + 16 + j * 16 + l15] = (__bf16)hv;  // h cols = 16..127
            }
        }
    }

    // MLP head (tiny): final h_T landed in buffer 0 (TLEN even). Lanes 0-15,
    // one batch row each.
    if (lane < 16) {
        const __bf16* hrow = Aw0 + lane * AROW + 16;
        float l1[50];
        for (int u = 0; u < 50; ++u) {
            float s = b1[u];
            for (int k = 0; k < HSZ; ++k) s += W1[u * HSZ + k] * (float)hrow[k];
            l1[u] = s > 0.0f ? s : 0.0f;
        }
        float l2[25];
        for (int u = 0; u < 25; ++u) {
            float s = b2[u];
            for (int k = 0; k < 50; ++k) s += W2[u * 50 + k] * l1[k];
            l2[u] = s > 0.0f ? s : 0.0f;
        }
        float s = b3[0];
        for (int k = 0; k < 25; ++k) s += W3[k] * l2[k];
        out[rowBase + lane] = s;
    }
}

extern "C" void kernel_launch(void* const* d_in, const int* in_sizes, int n_in,
                              void* d_out, int out_size, void* d_ws, size_t ws_size,
                              hipStream_t stream) {
    (void)in_sizes; (void)n_in; (void)out_size; (void)ws_size;
    const float* x   = (const float*)d_in[0];
    const float* Wih = (const float*)d_in[1];
    const float* Whh = (const float*)d_in[2];
    const float* bih = (const float*)d_in[3];
    const float* bhh = (const float*)d_in[4];
    const float* W1  = (const float*)d_in[5];
    const float* b1  = (const float*)d_in[6];
    const float* W2  = (const float*)d_in[7];
    const float* b2  = (const float*)d_in[8];
    const float* W3  = (const float*)d_in[9];
    const float* b3  = (const float*)d_in[10];

    __bf16* wsW = (__bf16*)d_ws;                 // 112*512 bf16 = 114688 B

    lstm_prep_kernel<<<224, 256, 0, stream>>>(Wih, Whh, bih, bhh, wsW);

    // 4096 rows / 16 rows-per-wave = 256 blocks of one wave each.
    lstm_main_kernel<<<256, 32, 0, stream>>>(x, wsW,
                                             W1, b1, W2, b2, W3, b3,
                                             (float*)d_out);
}
